// GroupAttention_35433480192574
// MI455X (gfx1250) — matrix operand order repaired and checked by reading, not compiled, not run on hardware
//
#include <hip/hip_runtime.h>
#include <hip/hip_bf16.h>
#include <math.h>

#define BB 4
#define SS 2048
#define DD 768
#define NCAT 1536          // wq|wk concatenated
#define MTOT (BB*SS)       // 8192

typedef _Float16 v16h __attribute__((ext_vector_type(16)));
typedef float    v8f  __attribute__((ext_vector_type(8)));
typedef float    v4f  __attribute__((ext_vector_type(4)));
typedef unsigned int v4u __attribute__((ext_vector_type(4)));

union FragU { v16h h; v4u u[2]; };

// ---------------- Kernel 0: LayerNorm + cast to f16 -----------------------
__global__ void ln_cast_kernel(const float* __restrict__ x,
                               const float* __restrict__ w,
                               const float* __restrict__ bias,
                               _Float16* __restrict__ out) {
  int row = blockIdx.x;                       // 0..MTOT-1
  int t = threadIdx.x;                        // 256 threads, D = 3*256
  const float* xr = x + (size_t)row * DD;
  float v0 = xr[t], v1 = xr[t + 256], v2 = xr[t + 512];
  float s = v0 + v1 + v2;
  float q = v0 * v0 + v1 * v1 + v2 * v2;
  __shared__ float sh1[256], sh2[256];
  sh1[t] = s; sh2[t] = q; __syncthreads();
  for (int off = 128; off > 0; off >>= 1) {
    if (t < off) { sh1[t] += sh1[t + off]; sh2[t] += sh2[t + off]; }
    __syncthreads();
  }
  float mu  = sh1[0] * (1.0f / DD);
  float var = sh2[0] * (1.0f / DD) - mu * mu;
  float inv = rsqrtf(var + 1e-12f);
  _Float16* o = out + (size_t)row * DD;
  o[t]       = (_Float16)((v0 - mu) * inv * w[t]       + bias[t]);
  o[t + 256] = (_Float16)((v1 - mu) * inv * w[t + 256] + bias[t + 256]);
  o[t + 512] = (_Float16)((v2 - mu) * inv * w[t + 512] + bias[t + 512]);
}

// ---------------- Kernel 0b: build transposed fused weight [NCAT, DD] f16 --
__global__ void wcat_kernel(const float* __restrict__ wq,
                            const float* __restrict__ wk,
                            _Float16* __restrict__ wt) {
  int idx = blockIdx.x * 256 + threadIdx.x;   // over NCAT*DD
  if (idx >= NCAT * DD) return;
  int n = idx / DD, k = idx % DD;
  float v = (n < DD) ? wq[(size_t)k * DD + n] : wk[(size_t)k * DD + (n - DD)];
  wt[(size_t)n * DD + k] = (_Float16)v;
}

// ---------------- Kernel 1: [q|k] = h @ [wq|wk] via WMMA ------------------
// Each wave computes a 64x64 tile: 4x4 grid of 16x16 accumulators,
// k-loop in steps of 32 using v_wmma_f32_16x16x32_f16.
// A frag as loaded here (contiguous k, not ISA table order): lane<16 holds row M=lane, K 0..15;
// lanes 16..31 hold same rows, K 16..31  -> two b128 loads/lane.
// B frag, same contiguous k as A: lane<16 holds col N=lane, K 0..15 (contig in
// transposed weight); lanes 16..31 hold K 16..31 -> two b128 loads/lane.
__global__ void __launch_bounds__(256)
qk_gemm_wmma(const _Float16* __restrict__ A,
             const _Float16* __restrict__ Bt,
             float* __restrict__ Cqk) {
  const int lane = threadIdx.x & 31;
  const int gw   = blockIdx.x * 8 + (threadIdx.x >> 5);
  const int m0   = (gw % (MTOT / 64)) * 64;
  const int n0   = (gw / (MTOT / 64)) * 64;
  const int l15  = lane & 15;
  const int hi   = lane >> 4;

  v8f acc[4][4] = {};
  const _Float16* aBase = A  + (size_t)(m0 + l15) * DD + hi * 16;
  const _Float16* bBase = Bt + (size_t)(n0 + l15) * DD + hi * 16;

  for (int k0 = 0; k0 < DD; k0 += 32) {
    FragU a[4], b[4];
#pragma unroll
    for (int i = 0; i < 4; ++i) {
      const _Float16* pa = aBase + (size_t)(i * 16) * DD + k0;
      a[i].u[0] = *(const v4u*)(pa);
      a[i].u[1] = *(const v4u*)(pa + 8);
      const _Float16* pb = bBase + (size_t)(i * 16) * DD + k0;
      b[i].u[0] = *(const v4u*)(pb);
      b[i].u[1] = *(const v4u*)(pb + 8);
      if (k0 + 32 < DD) {                      // hint next k-slice into cache
        __builtin_prefetch(pa + 32, 0, 1);
        __builtin_prefetch(pb + 32, 0, 1);
      }
    }
#pragma unroll
    for (int i = 0; i < 4; ++i)
#pragma unroll
      for (int j = 0; j < 4; ++j)
        acc[i][j] = __builtin_amdgcn_wmma_f32_16x16x32_f16(
            false, a[i].h, false, b[j].h, (short)0, acc[i][j], false, false);
  }

  // C/D layout: VGPR r -> row m0+i*16+r (lanes<16) / +8 (lanes>=16), col = lane&15
#pragma unroll
  for (int i = 0; i < 4; ++i)
#pragma unroll
    for (int j = 0; j < 4; ++j) {
      int col = n0 + j * 16 + l15;
#pragma unroll
      for (int r = 0; r < 8; ++r) {
        int row = m0 + i * 16 + r + hi * 8;
        Cqk[(size_t)row * NCAT + col] = acc[i][j][r];
      }
    }
}

// ---------------- Kernel 2: band scores + 2-entry softmax ----------------
__global__ void band_softmax_kernel(const float* __restrict__ QK,
                                    const float* __restrict__ bq,
                                    const float* __restrict__ bk,
                                    const int* __restrict__ amask,
                                    float* __restrict__ pup,
                                    float* __restrict__ pdn) {
  int widx = blockIdx.x * 8 + (threadIdx.x >> 5);   // one wave per (b,s)
  int lane = threadIdx.x & 31;
  int s = widx % SS;
  bool hasU = (s < SS - 1), hasD = (s > 0);
  const float* qr = QK + (size_t)widx * NCAT;
  const float* ku = QK + (size_t)(hasU ? widx + 1 : widx) * NCAT + DD;
  const float* kd = QK + (size_t)(hasD ? widx - 1 : widx) * NCAT + DD;
  float su = 0.f, sd = 0.f;
  for (int n = lane; n < DD; n += 32) {
    float qv = qr[n] + bq[n];
    float bkn = bk[n];
    if (hasU) su += qv * (ku[n] + bkn);
    if (hasD) sd += qv * (kd[n] + bkn);
  }
  for (int off = 16; off > 0; off >>= 1) {
    su += __shfl_xor(su, off, 32);
    sd += __shfl_xor(sd, off, 32);
  }
  if (lane == 0) {
    int b = widx / SS;
    const float invD = 1.0f / DD;
    su *= invD; sd *= invD;
    size_t base = (size_t)b * SS * SS + (size_t)s * SS;
    bool mu = hasU && ((amask[base + s + 1] & 1) != 0);
    bool md = hasD && ((amask[base + s - 1] & 1) != 0);
    float pu, pd;
    if (mu && md) {
      float m = fmaxf(su, sd);
      float eu = __expf(su - m), ed = __expf(sd - m);
      float inv = 1.0f / (eu + ed);
      pu = eu * inv; pd = ed * inv;
    } else if (mu) { pu = 1.f; pd = 0.f; }
    else if (md)   { pu = 0.f; pd = 1.f; }
    else           { pu = 0.f; pd = 0.f; }   // nan_to_num path
    pup[widx] = pu; pdn[widx] = pd;
  }
}

// ---------------- Kernel 3: symmetric band value, prior mix, logs ---------
__global__ void band_mix_kernel(const float* __restrict__ prior,
                                const float* __restrict__ pup,
                                const float* __restrict__ pdn,
                                float* __restrict__ nsup,
                                float* __restrict__ nsub,
                                float* __restrict__ Lr) {
  int idx = blockIdx.x * 256 + threadIdx.x;     // over B*(S-1)
  const int per = SS - 1;
  if (idx >= BB * per) return;
  int b = idx / per, s = idx % per;
  int w = b * SS + s;
  float v = sqrtf(pup[w] * pdn[w + 1] + 1e-9f);
  size_t basep = (size_t)b * SS * SS;
  float p1 = prior[basep + (size_t)s * SS + s + 1];
  float p2 = prior[basep + (size_t)(s + 1) * SS + s];
  float nu = p1 + (1.f - p1) * v;
  float nd = p2 + (1.f - p2) * v;
  nsup[w] = nu;
  nsub[w] = nd;
  Lr[w] = __logf(nu + 1e-9f);
}

// ---------------- Kernel 4: per-batch exclusive scan of L -----------------
__global__ void scan_kernel(const float* __restrict__ Lr, float* __restrict__ Cc) {
  int b = blockIdx.x, t = threadIdx.x;          // 256 threads, 8 elems each
  const float* L = Lr + (size_t)b * SS;
  float* C = Cc + (size_t)b * SS;
  int j0 = t * 8;
  float lv[8]; float part = 0.f;
#pragma unroll
  for (int e = 0; e < 8; ++e) {
    int j = j0 + e;
    lv[e] = (j < SS - 1) ? L[j] : 0.f;
    part += lv[e];
  }
  __shared__ float sp[256];
  sp[t] = part; __syncthreads();
  for (int off = 1; off < 256; off <<= 1) {
    float v = (t >= off) ? sp[t - off] : 0.f;
    __syncthreads();
    sp[t] += v;
    __syncthreads();
  }
  float acc = sp[t] - part;                     // exclusive prefix of partials
#pragma unroll
  for (int e = 0; e < 8; ++e) {
    int j = j0 + e;
    if (j < SS) C[j] = acc;                     // C[j] = sum_{x<j} L[x]
    acc += lv[e];
  }
}

// ---------------- Kernel 5: streaming output of g and n -------------------
__global__ void __launch_bounds__(256)
out_kernel(const float* __restrict__ prior,
           const float* __restrict__ nsup,
           const float* __restrict__ nsub,
           const float* __restrict__ Cc,
           float* __restrict__ gOut,
           float* __restrict__ nOut) {
  const float RT9 = 3.1622776601683794e-5f;     // sqrt(1e-9)
  int r = blockIdx.y;                           // b*S + i
  int b = r / SS, i = r % SS;
  int k0 = (blockIdx.x * blockDim.x + threadIdx.x) * 4;
  size_t rowOff = (size_t)r * SS + k0;
  v4f pr = *(const v4f*)(prior + rowOff);
  const float* Cb = Cc + (size_t)b * SS;
  v4f ck = *(const v4f*)(Cb + k0);
  float Ci = Cb[i];
  v4f gv, nv;
#pragma unroll
  for (int e = 0; e < 4; ++e) {
    int k = k0 + e;
    float p = pr[e];
    float n;
    if (k == i + 1)      n = nsup[b * SS + i];
    else if (k == i - 1) n = nsub[b * SS + k];
    else                 n = p + (1.f - p) * RT9;
    float g;
    if (k == i) {
      g = p + (1.f - p) * RT9;                  // diagonal carries n[i,i]
    } else {
      float d = (k > i) ? (ck[e] - Ci) : (Ci - ck[e]);
      g = __expf(d) + 1e-9f;
    }
    gv[e] = g; nv[e] = n;
  }
  __builtin_nontemporal_store(gv, (v4f*)(gOut + rowOff));
  __builtin_nontemporal_store(nv, (v4f*)(nOut + rowOff));
}

// --------------------------------------------------------------------------
extern "C" void kernel_launch(void* const* d_in, const int* in_sizes, int n_in,
                              void* d_out, int out_size, void* d_ws, size_t ws_size,
                              hipStream_t stream) {
  const float* hid   = (const float*)d_in[0];
  const int*   amask = (const int*)d_in[1];
  const float* prior = (const float*)d_in[2];
  const float* lnw   = (const float*)d_in[3];
  const float* lnb   = (const float*)d_in[4];
  const float* wq    = (const float*)d_in[5];
  const float* bq    = (const float*)d_in[6];
  const float* wk    = (const float*)d_in[7];
  const float* bk    = (const float*)d_in[8];

  char* ws = (char*)d_ws;
  size_t off = 0;
  auto carve = [&](size_t bytes) -> void* {
    void* p = ws + off;
    off = (off + bytes + 255) & ~(size_t)255;
    return p;
  };
  _Float16* hF  = (_Float16*)carve((size_t)MTOT * DD * sizeof(_Float16));
  _Float16* Wt  = (_Float16*)carve((size_t)NCAT * DD * sizeof(_Float16));
  float*    QK  = (float*)carve((size_t)MTOT * NCAT * sizeof(float));
  float*    pup = (float*)carve((size_t)MTOT * sizeof(float));
  float*    pdn = (float*)carve((size_t)MTOT * sizeof(float));
  float*    nsu = (float*)carve((size_t)MTOT * sizeof(float));
  float*    nsb = (float*)carve((size_t)MTOT * sizeof(float));
  float*    Lr  = (float*)carve((size_t)MTOT * sizeof(float));
  float*    Cc  = (float*)carve((size_t)MTOT * sizeof(float));

  float* gOut = (float*)d_out;
  float* nOut = gOut + (size_t)BB * SS * SS;

  ln_cast_kernel<<<MTOT, 256, 0, stream>>>(hid, lnw, lnb, hF);
  wcat_kernel<<<(NCAT * DD + 255) / 256, 256, 0, stream>>>(wq, wk, Wt);
  qk_gemm_wmma<<<(MTOT / 64) * (NCAT / 64) / 8, 256, 0, stream>>>(hF, Wt, QK);
  band_softmax_kernel<<<MTOT / 8, 256, 0, stream>>>(QK, bq, bk, amask, pup, pdn);
  band_mix_kernel<<<(BB * (SS - 1) + 255) / 256, 256, 0, stream>>>(prior, pup, pdn, nsu, nsb, Lr);
  scan_kernel<<<BB, 256, 0, stream>>>(Lr, Cc);
  out_kernel<<<dim3(SS / 1024, MTOT), 256, 0, stream>>>(prior, nsu, nsb, Cc, gOut, nOut);
}
